// ARTEncoder_67843303407662
// MI455X (gfx1250) — matrix operand order repaired and checked by reading, not compiled, not run on hardware
//
#include <hip/hip_runtime.h>
#include <hip/hip_bf16.h>
#include <cmath>

#define KOBJ 256
#define DDIM 512
#define HDIM 512
#define PDIM 128
#define EPAIR (KOBJ*(KOBJ-1))   // 65280
#define LN_EPS 1e-5f

typedef __bf16 bf16;
typedef __bf16 v16bf __attribute__((ext_vector_type(16)));
typedef __bf16 v8bf  __attribute__((ext_vector_type(8)));
typedef float  v8f   __attribute__((ext_vector_type(8)));

// ---------------------------------------------------------------------------
// B-operand swizzle (fragment-major): matrix stored as K-tiles of 32 rows x
// 16-col tiles, 512 bf16 per tile, lane L owns elements [L*16, L*16+16).
// For element (R, C) of the logical KxN matrix:
//   tile = (R>>5)*(N>>4) + (C>>4); lane = (C&15) + 16*((R>>3)&1); elem = (R&7) + 8*((R>>4)&1)
// A-operand fragments are two contiguous 16B runs per lane in row-major A.
// ---------------------------------------------------------------------------
__device__ __forceinline__ size_t bswz_idx(int R, int C, int N)
{
    size_t tile = (size_t)(R >> 5) * (N >> 4) + (C >> 4);
    int lane = (C & 15) + 16 * ((R >> 3) & 1);
    return tile * 512 + lane * 16 + ((R & 7) + 8 * ((R >> 4) & 1));
}

// ---------------------------------------------------------------------------
// Generic bf16 WMMA GEMM: out = act(A[MxKd] @ Bswz[KdxN] + bias)
// One wave per 16x16 tile; per-block N-span = (blockDim/32)*16.
// __syncthreads() between loads and stores makes A==out aliasing safe when
// one block covers the full N extent of its 16 rows (pf GEMM, blk=256).
// ---------------------------------------------------------------------------
__global__ void gemm_bf16_kernel(const bf16* __restrict__ A, const bf16* __restrict__ Bsw,
                                 const float* __restrict__ bias, int M, int N, int Kd,
                                 float* __restrict__ outF, bf16* __restrict__ outB,
                                 int relu, int swzOut)
{
    int lane = threadIdx.x & 31;
    int wave = threadIdx.x >> 5;
    int span = (blockDim.x >> 5) * 16;
    int col0 = blockIdx.x * span + wave * 16;
    int row0 = blockIdx.y * 16;

    int m  = lane & 15;          // A row within tile / C col within tile
    int kh = lane >> 4;          // K-half selector (M-half for C/D)
    int ntiles = N >> 4;

    v8f c = {0.f, 0.f, 0.f, 0.f, 0.f, 0.f, 0.f, 0.f};

    const bf16* arow = A + (size_t)(row0 + m) * Kd + 8 * kh;
    const bf16* bcol = Bsw + (size_t)(col0 >> 4) * 512 + (size_t)lane * 16;

    for (int k0 = 0; k0 < Kd; k0 += 32) {
        // A fragment: two contiguous 16B runs (ISA 16-bit A layout)
        const bf16* ap = arow + k0;
        v8bf alo = *(const v8bf*)ap;
        v8bf ahi = *(const v8bf*)(ap + 16);
        v16bf a = __builtin_shufflevector(alo, ahi,
                                          0,1,2,3,4,5,6,7,8,9,10,11,12,13,14,15);
        // B fragment: one aligned 32B load from swizzled layout
        const bf16* bp = bcol + (size_t)(k0 >> 5) * ntiles * 512;
        v16bf b = *(const v16bf*)bp;
        // prefetch next K-tile of B into cache (global_prefetch_b8)
        __builtin_prefetch(bp + (size_t)ntiles * 512, 0, 1);

        c = __builtin_amdgcn_wmma_f32_16x16x32_bf16(false, a, false, b,
                                                    (short)0, c, false, false);
    }

    __syncthreads();   // all A reads in block complete -> in-place store safe

    #pragma unroll
    for (int r = 0; r < 8; ++r) {
        int R = row0 + r + 8 * kh;
        int C = col0 + m;
        float v = c[r];
        if (bias) v += bias[C];
        if (relu) v = fmaxf(v, 0.f);
        if (outF) outF[(size_t)R * N + C] = v;
        if (outB) {
            size_t idx = swzOut ? bswz_idx(R, C, N) : ((size_t)R * N + C);
            outB[idx] = (bf16)v;
        }
    }
}

// ---------------------------------------------------------------------------
// Stage A: 13 geometric pair features -> relu(f @ w1 + b1), bf16 out (E x 128)
// ---------------------------------------------------------------------------
__global__ __launch_bounds__(256)
void pair_feats_kernel(const float* __restrict__ boxes,
                       const float* __restrict__ w1, const float* __restrict__ b1,
                       bf16* __restrict__ hOut)
{
    __shared__ float w1s[13 * PDIM];
    __shared__ float b1s[PDIM];
    int t = threadIdx.x;
    for (int i = t; i < 13 * PDIM; i += 256) w1s[i] = w1[i];
    if (t < PDIM) b1s[t] = b1[t];
    __syncthreads();

    int e = blockIdx.x * 256 + t;          // EPAIR == 255*256 exactly
    int i = e / 255;
    int r = e - i * 255;
    int j = r + (r >= i ? 1 : 0);

    float xi = boxes[i*4+0], yi = boxes[i*4+1], wi = boxes[i*4+2], hi = boxes[i*4+3];
    float xj = boxes[j*4+0], yj = boxes[j*4+1], wj = boxes[j*4+2], hj = boxes[j*4+3];

    float f[13];
    f[0] = (xj - xi) / wi;
    f[1] = (yj - yi) / hi;
    f[2] = logf(wj / wi + 1e-6f);
    f[3] = logf(hj / hi + 1e-6f);
    float iw = fmaxf(0.f, fminf(xi + wi, xj + wj) - fmaxf(xi, xj));
    float ih = fmaxf(0.f, fminf(yi + hi, yj + hj) - fmaxf(yi, yj));
    float inter = iw * ih;
    float iou = inter / (wi*hi + wj*hj - inter + 1e-6f);
    f[4] = wi; f[5] = hi; f[6] = wj; f[7] = hj; f[8] = iou;
    f[9] = xi; f[10] = yi; f[11] = xj; f[12] = yj;

    bf16* out = hOut + (size_t)e * PDIM;
    for (int p = 0; p < PDIM; ++p) {
        float acc = b1s[p];
        #pragma unroll
        for (int q = 0; q < 13; ++q) acc += f[q] * w1s[q*PDIM + p];
        out[p] = (bf16)fmaxf(acc, 0.f);
    }
}

// pf . att_w[2D:] for both layers (one wave per pair, shfl reduce)
__global__ __launch_bounds__(256)
void pfatt_kernel(const bf16* __restrict__ pf, const float* __restrict__ att1,
                  const float* __restrict__ att2,
                  float* __restrict__ out1, float* __restrict__ out2)
{
    int lane = threadIdx.x & 31;
    int wv   = threadIdx.x >> 5;
    int e = blockIdx.x * 8 + wv;
    const bf16* row = pf + (size_t)e * PDIM;
    float a1 = 0.f, a2 = 0.f;
    #pragma unroll
    for (int q = 0; q < PDIM/32; ++q) {
        int c = lane + 32*q;
        float v = (float)row[c];
        a1 += v * att1[c];
        a2 += v * att2[c];
    }
    for (int mk = 16; mk; mk >>= 1) {
        a1 += __shfl_xor(a1, mk, 32);
        a2 += __shfl_xor(a2, mk, 32);
    }
    if (lane == 0) { out1[e] = a1; out2[e] = a2; }
}

// xattA[i] = x[i].att_w[:D], xattB[i] = x[i].att_w[D:2D]
__global__ __launch_bounds__(256)
void xatt_kernel(const float* __restrict__ x, const float* __restrict__ attw,
                 float* __restrict__ outA, float* __restrict__ outB)
{
    int lane = threadIdx.x & 31;
    int wv   = threadIdx.x >> 5;
    int row = blockIdx.x * 8 + wv;
    const float* xr = x + (size_t)row * DDIM;
    float a = 0.f, b = 0.f;
    #pragma unroll
    for (int q = 0; q < DDIM/32; ++q) {
        int d = lane + 32*q;
        float v = xr[d];
        a += v * attw[d];
        b += v * attw[DDIM + d];
    }
    for (int mk = 16; mk; mk >>= 1) {
        a += __shfl_xor(a, mk, 32);
        b += __shfl_xor(b, mk, 32);
    }
    if (lane == 0) { outA[row] = a; outB[row] = b; }
}

// Per segment i: alpha row (K, diag 0) + apf = sum_j alpha*pf  -> Acat[i, 0:384]
// plus s_i = sum_j alpha.
__global__ __launch_bounds__(256)
void attn_reduce_kernel(const float* __restrict__ xattA, const float* __restrict__ xattB,
                        const float* __restrict__ pfatt, const float* __restrict__ attb,
                        const bf16* __restrict__ pf,
                        bf16* __restrict__ Acat, float* __restrict__ sVec)
{
    __shared__ float alphaS[KOBJ];
    __shared__ float part[256];
    int i = blockIdx.x;
    int t = threadIdx.x;

    int j = t;
    float a = 0.f;
    if (j != i) {
        int e = i*255 + j - (j > i ? 1 : 0);
        float lg = xattA[i] + xattB[j] + pfatt[e] + attb[0];
        a = 1.f / (1.f + expf(-lg));
    }
    alphaS[j] = a;
    Acat[(size_t)i * (KOBJ + PDIM) + j] = (bf16)a;
    __syncthreads();

    if (t < 32) {
        float p = 0.f;
        #pragma unroll
        for (int q = 0; q < 8; ++q) p += alphaS[t + 32*q];
        for (int mk = 16; mk; mk >>= 1) p += __shfl_xor(p, mk, 32);
        if (t == 0) sVec[i] = p;
    }

    // apf: 2 threads per column, 128 j's each; pf rows for segment i contiguous
    int c    = t & 127;
    int half = t >> 7;
    float acc = 0.f;
    const bf16* base = pf + (size_t)i * 255 * PDIM + c;
    for (int jj = half*128; jj < half*128 + 128; ++jj) {
        if (jj == i) continue;
        int e_off = jj - (jj > i ? 1 : 0);
        acc += alphaS[jj] * (float)base[(size_t)e_off * PDIM];
    }
    part[t] = acc;
    __syncthreads();
    if (t < 128)
        Acat[(size_t)i * (KOBJ + PDIM) + KOBJ + t] = (bf16)(part[t] + part[t + 128]);
}

// out = LN(base + (add + s_i*eb)*invden) * g + b ; H = 512, one block per row
__global__ __launch_bounds__(256)
void ln_kernel(const float* __restrict__ base, const float* __restrict__ add,
               const float* __restrict__ sVec, const float* __restrict__ eb,
               float invden, const float* __restrict__ g, const float* __restrict__ b,
               float* __restrict__ outF, bf16* __restrict__ outB)
{
    __shared__ float red[256];
    int row = blockIdx.x;
    int t = threadIdx.x;
    int c0 = t, c1 = t + 256;
    size_t o = (size_t)row * HDIM;
    float sb = sVec ? sVec[row] : 0.f;
    float v0 = base[o+c0] + (add[o+c0] + (eb ? sb*eb[c0] : 0.f)) * invden;
    float v1 = base[o+c1] + (add[o+c1] + (eb ? sb*eb[c1] : 0.f)) * invden;

    red[t] = v0 + v1;
    __syncthreads();
    for (int st = 128; st > 0; st >>= 1) { if (t < st) red[t] += red[t+st]; __syncthreads(); }
    float mu = red[0] * (1.f / HDIM);
    __syncthreads();
    float d0 = v0 - mu, d1 = v1 - mu;
    red[t] = d0*d0 + d1*d1;
    __syncthreads();
    for (int st = 128; st > 0; st >>= 1) { if (t < st) red[t] += red[t+st]; __syncthreads(); }
    float var = red[0] * (1.f / HDIM);
    float rs = rsqrtf(var + LN_EPS);

    float r0 = d0*rs*g[c0] + b[c0];
    float r1 = d1*rs*g[c1] + b[c1];
    outF[o+c0] = r0; outF[o+c1] = r1;
    if (outB) { outB[o+c0] = (bf16)r0; outB[o+c1] = (bf16)r1; }
}

// plain f32 -> bf16 (row-major, for A operands)
__global__ void f2bf_kernel(const float* __restrict__ src, bf16* __restrict__ dst, int n)
{
    int idx = blockIdx.x * blockDim.x + threadIdx.x;
    if (idx < n) dst[idx] = (bf16)src[idx];
}

// f32 -> bf16 with B-operand fragment swizzle; rowOff must be multiple of 32
__global__ void f2bf_swzB_kernel(const float* __restrict__ src, bf16* __restrict__ dst,
                                 int rows, int N, int rowOff)
{
    int idx = blockIdx.x * blockDim.x + threadIdx.x;
    if (idx >= rows * N) return;
    int r = idx / N, c = idx - r * N;
    dst[bswz_idx(r + rowOff, c, N)] = (bf16)src[idx];
}

// ---------------------------------------------------------------------------
extern "C" void kernel_launch(void* const* d_in, const int* in_sizes, int n_in,
                              void* d_out, int out_size, void* d_ws, size_t ws_size,
                              hipStream_t stream)
{
    (void)in_sizes; (void)n_in; (void)out_size; (void)ws_size;
    const float* x     = (const float*)d_in[0];
    const float* boxes = (const float*)d_in[1];
    const float* pe_w1 = (const float*)d_in[2];
    const float* pe_b1 = (const float*)d_in[3];
    const float* pe_w2 = (const float*)d_in[4];
    const float* pe_b2 = (const float*)d_in[5];
    const float* L[2][12];
    for (int l = 0; l < 2; ++l)
        for (int q = 0; q < 12; ++q)
            L[l][q] = (const float*)d_in[6 + l*12 + q];
    // q: 0 att_w, 1 att_b, 2 obj_w, 3 obj_b, 4 pair_w, 5 pair_b,
    //    6 ln_g, 7 ln_b, 8 f_w1, 9 f_b1, 10 f_w2, 11 f_b2

    char* ws = (char*)d_ws;
    size_t off = 0;
    auto alloc = [&](size_t bytes) -> char* {
        char* p = ws + off;
        off += (bytes + 255) & ~(size_t)255;
        return p;
    };
    bf16*  PF    = (bf16*)alloc((size_t)EPAIR * PDIM * 2);   // h, then pf in-place
    bf16*  PEW2B = (bf16*)alloc((size_t)PDIM * PDIM * 2);          // swizzled
    bf16*  XBF   = (bf16*)alloc((size_t)KOBJ * DDIM * 2);
    float* PFA1  = (float*)alloc((size_t)EPAIR * 4);
    float* PFA2  = (float*)alloc((size_t)EPAIR * 4);
    float* XATTA = (float*)alloc(KOBJ * 4);
    float* XATTB = (float*)alloc(KOBJ * 4);
    float* SV    = (float*)alloc(KOBJ * 4);
    bf16*  ACAT  = (bf16*)alloc((size_t)KOBJ * (KOBJ + PDIM) * 2);
    bf16*  BCAT  = (bf16*)alloc((size_t)(KOBJ + PDIM) * HDIM * 2); // swizzled
    float* MM    = (float*)alloc((size_t)KOBJ * HDIM * 4);
    float* OUT1F = (float*)alloc((size_t)KOBJ * HDIM * 4);
    bf16*  OUT1B = (bf16*)alloc((size_t)KOBJ * HDIM * 2);
    bf16*  TB    = (bf16*)alloc((size_t)KOBJ * HDIM * 2);
    float* FFN   = (float*)alloc((size_t)KOBJ * HDIM * 4);
    float* H1F   = (float*)alloc((size_t)KOBJ * HDIM * 4);
    bf16*  H1B   = (bf16*)alloc((size_t)KOBJ * HDIM * 2);
    bf16*  H2B   = (bf16*)alloc((size_t)KOBJ * HDIM * 2);
    bf16*  OBJWB = (bf16*)alloc((size_t)DDIM * HDIM * 2);          // swizzled
    bf16*  FW1B  = (bf16*)alloc((size_t)HDIM * HDIM * 2);          // swizzled
    bf16*  FW2B  = (bf16*)alloc((size_t)HDIM * HDIM * 2);          // swizzled

    auto cvt = [&](const float* s, bf16* d, int n) {
        f2bf_kernel<<<(n + 255) / 256, 256, 0, stream>>>(s, d, n);
    };
    auto cvtSwz = [&](const float* s, bf16* d, int rows, int N, int rowOff) {
        int n = rows * N;
        f2bf_swzB_kernel<<<(n + 255) / 256, 256, 0, stream>>>(s, d, rows, N, rowOff);
    };
    auto gemm = [&](const bf16* A, const bf16* B, const float* bias,
                    int M, int N, int Kd, float* oF, bf16* oB,
                    int relu, int blk, int swzOut) {
        int span = (blk / 32) * 16;
        dim3 grid(N / span, M / 16);
        gemm_bf16_kernel<<<grid, blk, 0, stream>>>(A, B, bias, M, N, Kd,
                                                   oF, oB, relu, swzOut);
    };

    // ---- Stage A: pair features (shared by both layers) ----
    cvt(x, XBF, KOBJ * DDIM);
    cvtSwz(pe_w2, PEW2B, PDIM, PDIM, 0);
    pair_feats_kernel<<<EPAIR / 256, 256, 0, stream>>>(boxes, pe_w1, pe_b1, PF);
    // pf = h @ pe_w2 + pe_b2, IN PLACE (block covers full N=128, barrier pre-store)
    gemm(PF, PEW2B, pe_b2, EPAIR, PDIM, PDIM, nullptr, PF, 0, 256, 0);
    pfatt_kernel<<<EPAIR / 8, 256, 0, stream>>>(PF, L[0][0] + 2*DDIM, L[1][0] + 2*DDIM,
                                                PFA1, PFA2);

    // ---- Two ART layers ----
    const float* curF = x;
    const bf16*  curB = XBF;
    for (int l = 0; l < 2; ++l) {
        const float* att_w = L[l][0]; const float* att_b = L[l][1];
        const float* obj_w = L[l][2]; const float* obj_b = L[l][3];
        const float* pair_w= L[l][4]; const float* pair_b= L[l][5];
        const float* ln_g  = L[l][6]; const float* ln_b  = L[l][7];
        const float* f_w1  = L[l][8]; const float* f_b1  = L[l][9];
        const float* f_w2  = L[l][10]; const float* f_b2 = L[l][11];
        float* pfa  = l ? PFA2 : PFA1;
        float* outF = l ? (float*)d_out : H1F;
        bf16*  outB = l ? H2B : H1B;

        cvtSwz(obj_w, OBJWB, DDIM, HDIM, 0);
        cvtSwz(pair_w, BCAT, PDIM, HDIM, KOBJ);   // Bcat K-rows 256..383
        cvtSwz(f_w1, FW1B, HDIM, HDIM, 0);
        cvtSwz(f_w2, FW2B, HDIM, HDIM, 0);

        xatt_kernel<<<KOBJ / 8, 256, 0, stream>>>(curF, att_w, XATTA, XATTB);
        // objx = x @ obj_w + obj_b  -> Bcat K-rows 0..255 (swizzled store)
        gemm(curB, OBJWB, obj_b, KOBJ, HDIM, DDIM, nullptr, BCAT, 0, 128, 1);
        attn_reduce_kernel<<<KOBJ, 256, 0, stream>>>(XATTA, XATTB, pfa, att_b, PF,
                                                     ACAT, SV);
        // mm = [Aalpha | apf] @ [objx ; pair_w]
        gemm(ACAT, BCAT, nullptr, KOBJ, HDIM, KOBJ + PDIM, MM, nullptr, 0, 128, 0);
        ln_kernel<<<KOBJ, 256, 0, stream>>>(curF, MM, SV, pair_b, 1.f / (KOBJ - 1),
                                            ln_g, ln_b, OUT1F, OUT1B);
        // FFN
        gemm(OUT1B, FW1B, f_b1, KOBJ, HDIM, HDIM, nullptr, TB, 1, 128, 0);
        gemm(TB, FW2B, f_b2, KOBJ, HDIM, HDIM, FFN, nullptr, 0, 128, 0);
        ln_kernel<<<KOBJ, 256, 0, stream>>>(OUT1F, FFN, nullptr, nullptr, 1.f,
                                            ln_g, ln_b, outF, outB);
        curF = outF; curB = outB;
    }
}
